// CrossModalDeformableAttentionLite_62878321214196
// MI455X (gfx1250) — matrix operand order repaired and checked by reading, not compiled, not run on hardware
//
#include <hip/hip_runtime.h>
#include <hip/hip_bf16.h>
#include <math.h>

typedef __attribute__((ext_vector_type(16))) __bf16 v16bf;
typedef __attribute__((ext_vector_type(8)))  float  v8f;

__device__ __forceinline__ unsigned short f32_to_bf16(float f) {
    unsigned int u = __builtin_bit_cast(unsigned int, f);
    u += 0x7FFFu + ((u >> 16) & 1u);           // round-to-nearest-even
    return (unsigned short)(u >> 16);
}
__device__ __forceinline__ float bf16_to_f32(unsigned short h) {
    unsigned int u = ((unsigned int)h) << 16;
    return __builtin_bit_cast(float, u);
}

// ---------------------------------------------------------------------------
// Generic bf16-WMMA GEMM:  Out[m,n] = act(mask * (sum_k A[m,k]*W[n,k] + bias[n]))
//   TIN     : float (converted via VGPRs) or unsigned short (bf16 bits;
//             A-tile staged with global_load_async_to_lds_b128, ASYNCcnt path)
//   ATRANS  : A stored [K,M] (k-major) instead of [M,K]
//   RELU    : apply ReLU in epilogue
//   MASK    : multiply row by mask[row] (after bias, like reference)
//   OUT_F32 : write fp32 instead of bf16
// Workgroup = 128 threads (4 waves), tile 128(M) x 64(N), K step 32.
// Each wave owns a 32x64 strip -> 8 v_wmma_f32_16x16x32_bf16 per K step
// (2 A fragments amortized over 4 staged B tiles).
// ---------------------------------------------------------------------------
template <typename TIN, bool ATRANS, bool RELU, bool MASK, bool OUT_F32>
__global__ __launch_bounds__(128)
void gemm_wmma_bf16(const TIN* __restrict__ A, int strideAK,
                    const float* __restrict__ Wt,    // [N,K] (torch layout)
                    const float* __restrict__ bias,  // [N]
                    const int*  __restrict__ mask,   // [M] or nullptr
                    void* __restrict__ Out, int ldout,
                    int K,
                    long aBatchStride, long oBatchStride)
{
    __shared__ __align__(16) unsigned short As[128][40]; // [m][k], pad kills bank conflicts
    __shared__ __align__(16) unsigned short Bs[64][40];  // [n][k]

    const int tid  = threadIdx.x;
    const int wave = tid >> 5;
    const int lane = tid & 31;
    const int m0   = blockIdx.x * 128;
    const int n0   = blockIdx.y * 64;
    const int b    = blockIdx.z;

    const TIN* __restrict__ Ab = A + (long)b * aBatchStride;

    v8f acc[2][4];
    const v8f vzero = {0.f,0.f,0.f,0.f,0.f,0.f,0.f,0.f};
    #pragma unroll
    for (int i = 0; i < 2; ++i)
        #pragma unroll
        for (int j = 0; j < 4; ++j) acc[i][j] = vzero;

    const int lm = lane & 15;   // M-row (A) / N-col (B) within 16
    const int kh = lane >> 4;   // K-half selector

    for (int kb = 0; kb < K; kb += 32) {
        // ---- stage A tile (128 x 32): one row (64B) per thread ----
        if constexpr (sizeof(TIN) == 2 && !ATRANS) {
            // activation already bf16: DMA straight into LDS (ASYNCcnt path)
            const TIN* src = Ab + (long)(m0 + tid) * K + kb;
            const unsigned lds0 = (unsigned)(unsigned long long)&As[tid][0];
            const unsigned long long g0 = (unsigned long long)src;
            asm volatile("global_load_async_to_lds_b128 %0, %1, off"
                         :: "v"(lds0), "v"(g0) : "memory");
            asm volatile("global_load_async_to_lds_b128 %0, %1, off offset:16"
                         :: "v"(lds0), "v"(g0) : "memory");
            asm volatile("global_load_async_to_lds_b128 %0, %1, off offset:32"
                         :: "v"(lds0), "v"(g0) : "memory");
            asm volatile("global_load_async_to_lds_b128 %0, %1, off offset:48"
                         :: "v"(lds0), "v"(g0) : "memory");
        } else if constexpr (!ATRANS) {
            const float* src = (const float*)Ab + (long)(m0 + tid) * K + kb;
            __builtin_prefetch(src + 32, 0, 1);          // next K tile
            alignas(16) unsigned short tmp[32];
            const float4* s4 = (const float4*)src;
            #pragma unroll
            for (int i = 0; i < 8; ++i) {
                float4 v = s4[i];
                tmp[i*4+0] = f32_to_bf16(v.x);
                tmp[i*4+1] = f32_to_bf16(v.y);
                tmp[i*4+2] = f32_to_bf16(v.z);
                tmp[i*4+3] = f32_to_bf16(v.w);
            }
            #pragma unroll
            for (int i = 0; i < 4; ++i)
                *(uint4*)&As[tid][i*8] = ((const uint4*)tmp)[i];
        } else {
            // A is [K, M] (feature map): contiguous along M
            const int j    = tid & 31;          // k within step
            const int mseg = tid >> 5;          // 0..3 -> 32 m each
            const float* src = (const float*)Ab + (long)(kb + j) * strideAK + (m0 + (mseg << 5));
            const float4* s4 = (const float4*)src;
            #pragma unroll
            for (int i2 = 0; i2 < 8; ++i2) {
                float4 v = s4[i2];
                As[(mseg << 5) + i2*4 + 0][j] = f32_to_bf16(v.x);
                As[(mseg << 5) + i2*4 + 1][j] = f32_to_bf16(v.y);
                As[(mseg << 5) + i2*4 + 2][j] = f32_to_bf16(v.z);
                As[(mseg << 5) + i2*4 + 3][j] = f32_to_bf16(v.w);
            }
        }
        // ---- stage B tile: W[n0..n0+63, kb..kb+31] (contiguous along k) ----
        {
            const int n    = tid >> 1;
            const int koff = (tid & 1) << 4;
            const float* src = Wt + (long)(n0 + n) * K + (kb + koff);
            alignas(16) unsigned short tmp[16];
            const float4* s4 = (const float4*)src;
            #pragma unroll
            for (int i = 0; i < 4; ++i) {
                float4 v = s4[i];
                tmp[i*4+0] = f32_to_bf16(v.x);
                tmp[i*4+1] = f32_to_bf16(v.y);
                tmp[i*4+2] = f32_to_bf16(v.z);
                tmp[i*4+3] = f32_to_bf16(v.w);
            }
            *(uint4*)&Bs[n][koff]     = ((const uint4*)tmp)[0];
            *(uint4*)&Bs[n][koff + 8] = ((const uint4*)tmp)[1];
        }
        if constexpr (sizeof(TIN) == 2 && !ATRANS) {
            asm volatile("s_wait_asynccnt 0x0" ::: "memory");
        }
        __syncthreads();

        // ---- preload ALL fragments (12x ds_load_b128), then 8 WMMAs ----
        union Frag { v16bf v; unsigned int u[8]; };
        Frag fa[2];
        #pragma unroll
        for (int i = 0; i < 2; ++i) {
            #pragma unroll
            for (int v = 0; v < 8; ++v) {
                // A frag (16x32, 16-bit): lanes0-15 K={0..7,16..23}, lanes16-31 K={8..15,24..31}
                const int k = ((v < 4) ? (2 * v) : (16 + 2 * (v - 4))) + 8 * kh;
                fa[i].u[v] = *(const unsigned int*)&As[(wave << 5) + (i << 4) + lm][k];
            }
        }
        Frag fb[4];
        #pragma unroll
        for (int j = 0; j < 4; ++j) {
            #pragma unroll
            for (int v = 0; v < 8; ++v) {
                // B frag (32x16): lanes0-15 K={0..7,16..23}, lanes16-31 K={8..15,24..31}
                const int k = ((v < 4) ? (2 * v) : (16 + 2 * (v - 4))) + 8 * kh;
                fb[j].u[v] = *(const unsigned int*)&Bs[(j << 4) + lm][k];
            }
        }
        #pragma unroll
        for (int i = 0; i < 2; ++i)
            #pragma unroll
            for (int j = 0; j < 4; ++j)
                acc[i][j] = __builtin_amdgcn_wmma_f32_16x16x32_bf16(
                                false, fa[i].v, false, fb[j].v, (short)0, acc[i][j], false, false);
        __syncthreads();
    }

    // ---- epilogue: C/D layout lane=N-col, VGPR r -> M = r + 8*(lane>>4)
    const int colL  = lane & 15;
    const int rbase = (lane >> 4) * 8;
    #pragma unroll
    for (int i = 0; i < 2; ++i) {
        #pragma unroll
        for (int j = 0; j < 4; ++j) {
            const int col = n0 + (j << 4) + colL;
            const float bv = bias[col];
            #pragma unroll
            for (int r = 0; r < 8; ++r) {
                const long row = m0 + (wave << 5) + (i << 4) + rbase + r;
                float val = acc[i][j][r] + bv;
                if (MASK) val *= (float)mask[row];
                if (RELU) val = fmaxf(val, 0.f);
                const long oi = (long)b * oBatchStride + row * (long)ldout + col;
                if (OUT_F32) ((float*)Out)[oi] = val;
                else         ((unsigned short*)Out)[oi] = f32_to_bf16(val);
            }
        }
    }
}

// ---------------------------------------------------------------------------
// One wave per query: 12 dot products (offsets+weights), tanh/softmax/mask,
// 4 samples x 4 bilinear corners gathered from pf [B,HW,256] bf16,
// weighted context written to hbuf[:, 256:512] as bf16. Lane owns 8 channels.
// ---------------------------------------------------------------------------
__global__ __launch_bounds__(256)
void deform_sample_kernel(const float* __restrict__ queries,    // [B*Q,256]
                          const float* __restrict__ ref_points, // [B*Q,2]
                          const int*  __restrict__ valid_mask,  // [B*Q]
                          const float* __restrict__ Wo, const float* __restrict__ bo,
                          const float* __restrict__ Ww, const float* __restrict__ bw,
                          const unsigned short* __restrict__ pf, // [B,4096,256] bf16
                          unsigned short* __restrict__ hbuf,     // [B*Q,512]
                          int Q)
{
    const int lane = threadIdx.x & 31;
    const int row  = blockIdx.x * (blockDim.x >> 5) + (threadIdx.x >> 5);
    const int b    = row / Q;

    const float* qrow = queries + (long)row * 256 + lane * 8;
    const float4 q0 = *(const float4*)qrow;
    const float4 q1 = *(const float4*)(qrow + 4);

    float dots[12];
    #pragma unroll
    for (int r = 0; r < 12; ++r) {
        const float* wrow = ((r < 8) ? (Wo + r * 256) : (Ww + (r - 8) * 256)) + lane * 8;
        const float4 w0 = *(const float4*)wrow;
        const float4 w1 = *(const float4*)(wrow + 4);
        float p = q0.x*w0.x + q0.y*w0.y + q0.z*w0.z + q0.w*w0.w
                + q1.x*w1.x + q1.y*w1.y + q1.z*w1.z + q1.w*w1.w;
        #pragma unroll
        for (int s = 16; s >= 1; s >>= 1) p += __shfl_xor(p, s, 32);
        dots[r] = p;  // full sum in every lane
    }

    // softmax over S=4 (mask==0 -> all weights 0 -> context 0, matches reference)
    float wr[4], wmax = -1e30f;
    #pragma unroll
    for (int s = 0; s < 4; ++s) { wr[s] = dots[8 + s] + bw[s]; wmax = fmaxf(wmax, wr[s]); }
    float denom = 0.f;
    #pragma unroll
    for (int s = 0; s < 4; ++s) { wr[s] = __expf(wr[s] - wmax); denom += wr[s]; }

    const float m  = (float)valid_mask[row];
    const float rx = ref_points[(long)row * 2 + 0];
    const float ry = ref_points[(long)row * 2 + 1];

    float ctx[8] = {0.f,0.f,0.f,0.f,0.f,0.f,0.f,0.f};
    if (m != 0.f) {
        #pragma unroll
        for (int s = 0; s < 4; ++s) {
            const float ws = wr[s] / denom;
            const float ox = tanhf(dots[2*s]     + bo[2*s])     * 0.05f;
            const float oy = tanhf(dots[2*s + 1] + bo[2*s + 1]) * 0.05f;
            const float gx = fminf(fmaxf(rx + ox, -1.2f), 1.2f);
            const float gy = fminf(fmaxf(ry + oy, -1.2f), 1.2f);
            const float x  = (gx + 1.f) * 0.5f * 63.f;
            const float y  = (gy + 1.f) * 0.5f * 63.f;
            const float x0f = floorf(x), y0f = floorf(y);
            const int   x0 = (int)x0f,  y0 = (int)y0f;
            const float wx1 = x - x0f, wx0 = 1.f - wx1;
            const float wy1 = y - y0f, wy0 = 1.f - wy1;
            #pragma unroll
            for (int c = 0; c < 4; ++c) {
                const int xi = x0 + (c & 1), yi = y0 + (c >> 1);
                const float wgt = ((c & 1) ? wx1 : wx0) * ((c >> 1) ? wy1 : wy0) * ws;
                if (xi >= 0 && xi < 64 && yi >= 0 && yi < 64) {  // zeros padding
                    const unsigned short* prow =
                        pf + (((long)b * 4096 + yi * 64 + xi) * 256 + lane * 8);
                    const uint4 pk = *(const uint4*)prow;
                    const unsigned int pu[4] = {pk.x, pk.y, pk.z, pk.w};
                    #pragma unroll
                    for (int i = 0; i < 4; ++i) {
                        ctx[2*i]   += wgt * bf16_to_f32((unsigned short)(pu[i] & 0xFFFFu));
                        ctx[2*i+1] += wgt * bf16_to_f32((unsigned short)(pu[i] >> 16));
                    }
                }
            }
        }
    }

    unsigned int op[4];
    #pragma unroll
    for (int i = 0; i < 4; ++i)
        op[i] = (unsigned int)f32_to_bf16(ctx[2*i]) |
                ((unsigned int)f32_to_bf16(ctx[2*i+1]) << 16);
    uint4 ov; ov.x = op[0]; ov.y = op[1]; ov.z = op[2]; ov.w = op[3];
    *(uint4*)(hbuf + (long)row * 512 + 256 + lane * 8) = ov;
}

// ---------------------------------------------------------------------------
extern "C" void kernel_launch(void* const* d_in, const int* in_sizes, int n_in,
                              void* d_out, int out_size, void* d_ws, size_t ws_size,
                              hipStream_t stream) {
    const float* queries = (const float*)d_in[0];
    const float* refpts  = (const float*)d_in[1];
    const float* feat    = (const float*)d_in[2];
    const int*   vmask   = (const int*)  d_in[3];
    const float* Wq = (const float*)d_in[4];
    const float* bq = (const float*)d_in[5];
    const float* Wf = (const float*)d_in[6];
    const float* bf = (const float*)d_in[7];
    const float* Wo = (const float*)d_in[8];
    const float* bo = (const float*)d_in[9];
    const float* Ww = (const float*)d_in[10];
    const float* bw = (const float*)d_in[11];
    const float* W1 = (const float*)d_in[12];
    const float* b1 = (const float*)d_in[13];
    const float* W2 = (const float*)d_in[14];
    const float* b2 = (const float*)d_in[15];

    const int B = 8, Q = 8192, D = 256, C = 256, H = 64, W = 64, Hd = 256;
    const int HW = H * W;
    const int BQ = B * Q;                 // 65536

    char* ws = (char*)d_ws;
    unsigned short* pf   = (unsigned short*)ws;                                    // B*HW*Hd bf16 (16.8 MB)
    unsigned short* hbuf = (unsigned short*)(ws + (size_t)B * HW * Hd * 2);        // BQ*512 bf16 (67 MB)
    unsigned short* h1   = (unsigned short*)(ws + (size_t)B * HW * Hd * 2
                                                + (size_t)BQ * 512 * 2);           // BQ*Hd bf16 (33.5 MB)

    const dim3 blk(128);

    // 1) pf[b, pixel, hd] = Wf . feat[b]  (A transposed: feat is [C, HW])
    gemm_wmma_bf16<float, true, false, false, false>
        <<<dim3(HW / 128, Hd / 64, B), blk, 0, stream>>>(
            feat, HW, Wf, bf, nullptr, pf, Hd, C, (long)C * HW, (long)HW * Hd);

    // 2) pq (masked) -> hbuf[:, 0:256]
    gemm_wmma_bf16<float, false, false, true, false>
        <<<dim3(BQ / 128, Hd / 64, 1), blk, 0, stream>>>(
            queries, 0, Wq, bq, vmask, hbuf, 512, D, 0, 0);

    // 3) deformable sampling -> hbuf[:, 256:512]
    deform_sample_kernel<<<dim3(BQ / 8), 256, 0, stream>>>(
        queries, refpts, vmask, Wo, bo, Ww, bw, pf, hbuf, Q);

    // 4) h1 = relu(hbuf . W1^T + b1)   (K = 512, bf16 A staged via async-to-LDS)
    gemm_wmma_bf16<unsigned short, false, true, false, false>
        <<<dim3(BQ / 128, Hd / 64, 1), blk, 0, stream>>>(
            hbuf, 0, W1, b1, nullptr, h1, Hd, 2 * Hd, 0, 0);

    // 5) out = h1 . W2^T + b2  (fp32 -> d_out, bf16 A staged via async-to-LDS)
    gemm_wmma_bf16<unsigned short, false, false, false, true>
        <<<dim3(BQ / 128, Hd / 64, 1), blk, 0, stream>>>(
            h1, 0, W2, b2, nullptr, d_out, Hd, Hd, 0, 0);
}
